// HebbianLinear_43121471652136
// MI455X (gfx1250) — compile-verified, matrix operand order checked by reading, not run on hardware
//
#include <hip/hip_runtime.h>

// MI455X / gfx1250: wave32, WMMA 16x16x32 bf16 -> f32 accum.
// Pipeline: prep (convert/transpose/colsum) -> ym matvec ->
//   GEMM1 (yT = (x@W^T)^T, bf16) -> GEMM2+update (G^T = xT@y_, fused -> new_W bf16)
//   -> GEMM3 (out = x@new_W^T + bias, f32).
// GEMM: 256x128 block tile, 8 waves of 64x64, K-step 64 (2 slabs of 32),
// 32 wmma per barrier pair, double-buffered LDS filled with async
// global->LDS (ASYNCcnt), steady-state loop peeled (no per-iter branches).

typedef __bf16 bf16;
typedef __attribute__((ext_vector_type(8)))  __bf16 v8bf;
typedef __attribute__((ext_vector_type(16))) __bf16 v16bf;
typedef __attribute__((ext_vector_type(8)))  float  v8f;
typedef __attribute__((ext_vector_type(4)))  float  v4f;
typedef __attribute__((ext_vector_type(4)))  int    v4i;

#define LDSS 40  // LDS slab row stride (bf16 elems): 80B rows, conflict-free b128 reads

#if defined(__has_builtin)
#if __has_builtin(__builtin_amdgcn_global_load_async_to_lds_b128)
#define HAS_ASYNC 1
#endif
#endif
#ifndef HAS_ASYNC
#define HAS_ASYNC 0
#endif

#if HAS_ASYNC
#define AS1 __attribute__((address_space(1)))
#define AS3 __attribute__((address_space(3)))
__device__ __forceinline__ void async_ld128(const bf16* g, bf16* l) {
  __builtin_amdgcn_global_load_async_to_lds_b128((AS1 v4i*)(g), (AS3 v4i*)(l),
                                                 0, 0);
}
#if __has_builtin(__builtin_amdgcn_s_wait_asynccnt)
#define WAIT_ASYNC(n) __builtin_amdgcn_s_wait_asynccnt(n)
#else
#define WAIT_ASYNC(n) asm volatile("s_wait_asynccnt %0" ::"n"(n) : "memory")
#endif
#endif

__global__ void k_zero(float* p, int n) {
  int i = blockIdx.x * blockDim.x + threadIdx.x;
  if (i < n) p[i] = 0.f;
}

// x [4096,2048] f32 -> xbf (bf16, same layout) and xT [2048,4096] bf16
__global__ void k_xprep(const float* __restrict__ x, bf16* __restrict__ xbf,
                        bf16* __restrict__ xT) {
  __shared__ float t[32][33];
  const int tx = threadIdx.x, ty = threadIdx.y;
  const int colBase = blockIdx.x * 32, rowBase = blockIdx.y * 32;
#pragma unroll
  for (int s = 0; s < 4; ++s) {
    int r = ty + 8 * s;
    float v = x[(size_t)(rowBase + r) * 2048 + colBase + tx];
    xbf[(size_t)(rowBase + r) * 2048 + colBase + tx] = (bf16)v;
    t[r][tx] = v;
  }
  __syncthreads();
#pragma unroll
  for (int s = 0; s < 4; ++s) {
    int r = ty + 8 * s;
    xT[(size_t)(colBase + r) * 4096 + rowBase + tx] = (bf16)t[tx][r];
  }
}

// column sums of x into xm (pre-zeroed); grid (8,32), block 256
__global__ void k_colsum(const float* __restrict__ x, float* __restrict__ xm) {
  const int col = blockIdx.x * 256 + threadIdx.x;
  const int r0 = blockIdx.y * 128;
  float s = 0.f;
  for (int r = 0; r < 128; ++r) s += x[(size_t)(r0 + r) * 2048 + col];
  atomicAdd(&xm[col], s);
}

// f32 -> bf16, 8 elements/thread
__global__ void k_f2bf(const float* __restrict__ src, bf16* __restrict__ dst) {
  size_t i = ((size_t)blockIdx.x * 256 + threadIdx.x) * 8;
  v4f a = *(const v4f*)(src + i);
  v4f b = *(const v4f*)(src + i + 4);
  v8bf o;
#pragma unroll
  for (int v = 0; v < 4; ++v) { o[v] = (bf16)a[v]; o[v + 4] = (bf16)b[v]; }
  *(v8bf*)(dst + i) = o;
}

// ym[o] = dot(W[o,:], xm_sum) / B   (== mean_b y_[b,o] exactly)
__global__ void k_ym(const float* __restrict__ W, const float* __restrict__ xm,
                     float* __restrict__ ym) {
  __shared__ float red[256];
  const int o = blockIdx.x, t = threadIdx.x;
  float s = 0.f;
  for (int i = t; i < 2048; i += 256) s += W[(size_t)o * 2048 + i] * xm[i];
  red[t] = s;
  __syncthreads();
  for (int w = 128; w > 0; w >>= 1) {
    if (t < w) red[t] += red[t + w];
    __syncthreads();
  }
  if (t == 0) ym[o] = red[0] * (1.f / 4096.f);
}

// One 16x32 bf16 operand fragment from an LDS slab (row-major, stride LDSS).
// ISA 16-bit A layout: lane (l&15)=row, K-half = (l>>4)*8 -> {kh..kh+7, kh+16..kh+23}.
__device__ __forceinline__ v16bf ldfrag(const bf16* p) {
  v8bf lo = *(const v8bf*)p;
  v8bf hi = *(const v8bf*)(p + 16);
  return __builtin_shufflevector(lo, hi, 0, 1, 2, 3, 4, 5, 6, 7, 8, 9, 10, 11,
                                 12, 13, 14, 15);
}

// C = A(MxK) * Bop(NxK)^T with bf16 operands, f32 accum. Block 256x128.
// MODE 0: store C^T as bf16 (ld = M)          [GEMM1 -> yT]
// MODE 1: fused Hebbian update -> new_W bf16  [C[m,n] = G[n,m], M==IN, N==OUT]
// MODE 2: outF[m*N+n] = C + bias[n]           [GEMM3 -> y]
template <int MODE>
__global__ void __launch_bounds__(256) k_gemm(
    const bf16* __restrict__ A, const bf16* __restrict__ Bm, int M, int N,
    int K, bf16* __restrict__ outT, const float* __restrict__ W,
    const float* __restrict__ wa, const float* __restrict__ wb,
    const float* __restrict__ wc, const float* __restrict__ wd,
    const float* __restrict__ xm, const float* __restrict__ ym,
    bf16* __restrict__ nwb, const float* __restrict__ bias,
    float* __restrict__ outF) {
  // Double-buffered 64-wide K tiles as 2 slabs of 32: ~120 KB of LDS (WGP has 320 KB).
  __shared__ bf16 As[2][2][256 * LDSS];
  __shared__ bf16 Bs[2][2][128 * LDSS];
  const int tid = threadIdx.x;
  const int lane = tid & 31, wave = tid >> 5;
  const int wm = (wave >> 1) * 64;  // wave M offset: 0/64/128/192
  const int wn = (wave & 1) * 64;   // wave N offset: 0/64
  const int lr = lane & 15;
  const int kh = (lane >> 4) * 8;
  const int mBase = blockIdx.y * 256;
  const int nBase = blockIdx.x * 128;

  v8f acc[4][4];
#pragma unroll
  for (int i = 0; i < 4; ++i)
#pragma unroll
    for (int j = 0; j < 4; ++j) acc[i][j] = v8f{};

  // 32 wmma over a 64-wide K tile (two slabs)
  auto compute64 = [&](int buf) {
#pragma unroll
    for (int s = 0; s < 2; ++s) {
      v16bf af[4], bfr[4];
#pragma unroll
      for (int i = 0; i < 4; ++i)
        af[i] = ldfrag(&As[buf][s][(wm + i * 16 + lr) * LDSS + kh]);
#pragma unroll
      for (int j = 0; j < 4; ++j)
        bfr[j] = ldfrag(&Bs[buf][s][(wn + j * 16 + lr) * LDSS + kh]);
#pragma unroll
      for (int i = 0; i < 4; ++i)
#pragma unroll
        for (int j = 0; j < 4; ++j)
          acc[i][j] = __builtin_amdgcn_wmma_f32_16x16x32_bf16(
              false, af[i], false, bfr[j], (short)0, acc[i][j], false, false);
    }
  };

#if HAS_ASYNC
  // 12 async b128 per thread per 256x64 (A) + 128x64 (B) tile.
  auto issue = [&](int k0, int buf) {
#pragma unroll
    for (int p = 0; p < 8; ++p) {
      int chunk = tid + p * 256;  // 2048 chunks cover A tile (256 rows x 64)
      int row = chunk >> 3, c = (chunk & 7) * 8;
      async_ld128(&A[(size_t)(mBase + row) * K + k0 + c],
                  &As[buf][c >> 5][row * LDSS + (c & 31)]);
    }
#pragma unroll
    for (int p = 0; p < 4; ++p) {
      int chunk = tid + p * 256;  // 1024 chunks cover B tile (128 rows x 64)
      int row = chunk >> 3, c = (chunk & 7) * 8;
      async_ld128(&Bm[(size_t)(nBase + row) * K + k0 + c],
                  &Bs[buf][c >> 5][row * LDSS + (c & 31)]);
    }
  };
  const int nT = K >> 6;  // number of 64-wide K tiles
  issue(0, 0);
  for (int t = 0; t + 1 < nT; ++t) {  // steady state: branch-free body
    issue((t + 1) << 6, (t + 1) & 1);  // prefetch next tile
    WAIT_ASYNC(12);                    // oldest 12 (current tile) complete
    __syncthreads();
    compute64(t & 1);
    __syncthreads();  // all waves done with buf before it is refilled
  }
  WAIT_ASYNC(0);  // peeled last tile
  __syncthreads();
  compute64((nT - 1) & 1);
#else
  for (int k0 = 0; k0 < K; k0 += 64) {
    __syncthreads();
#pragma unroll
    for (int p = 0; p < 8; ++p) {
      int chunk = tid + p * 256;
      int row = chunk >> 3, c = (chunk & 7) * 8;
      *(v8bf*)&As[0][c >> 5][row * LDSS + (c & 31)] =
          *(const v8bf*)&A[(size_t)(mBase + row) * K + k0 + c];
    }
#pragma unroll
    for (int p = 0; p < 4; ++p) {
      int chunk = tid + p * 256;
      int row = chunk >> 3, c = (chunk & 7) * 8;
      *(v8bf*)&Bs[0][c >> 5][row * LDSS + (c & 31)] =
          *(const v8bf*)&Bm[(size_t)(nBase + row) * K + k0 + c];
    }
    __syncthreads();
    compute64(0);
  }
#endif

  // Epilogue. f32 C layout: lane -> N = lane&15 (fixed), M = v + 8*(lane>>4).
  const int nOff = lane & 15;
  const int m8 = 8 * (lane >> 4);
#pragma unroll
  for (int i = 0; i < 4; ++i) {
#pragma unroll
    for (int j = 0; j < 4; ++j) {
      const int ng = nBase + wn + j * 16 + nOff;
      const int mg = mBase + wm + i * 16 + m8;
      if (MODE == 0) {
        v8bf o;
#pragma unroll
        for (int v = 0; v < 8; ++v) o[v] = (bf16)acc[i][j][v];
        *(v8bf*)&outT[(size_t)ng * M + mg] = o;  // yT[n, m..m+7]
      } else if (MODE == 1) {
        const float ETA = 0.001f, INVB = 1.f / 4096.f;
        const size_t base = (size_t)ng * M + mg;  // new_W[o, i..i+7], M==IN
        const float yo = ym[ng];
        v8bf o;
#pragma unroll
        for (int v = 0; v < 8; ++v) {
          float g = acc[i][j][v];  // G[o, i+v]
          float dw = xm[mg + v] * INVB * wa[base + v] + yo * wb[base + v] +
                     g * INVB * wc[base + v] + wd[base + v];
          o[v] = (bf16)((1.f - ETA) * W[base + v] + ETA * dw);
        }
        *(v8bf*)&nwb[base] = o;
      } else {
        const float bi = bias[ng];
#pragma unroll
        for (int v = 0; v < 8; ++v)
          outF[(size_t)(mg + v) * N + ng] = acc[i][j][v] + bi;
      }
    }
  }
}

extern "C" void kernel_launch(void* const* d_in, const int* in_sizes, int n_in,
                              void* d_out, int out_size, void* d_ws,
                              size_t ws_size, hipStream_t stream) {
  (void)in_sizes; (void)n_in; (void)out_size; (void)ws_size;
  const float* x  = (const float*)d_in[0];
  const float* wa = (const float*)d_in[1];
  const float* wb = (const float*)d_in[2];
  const float* wc = (const float*)d_in[3];
  const float* wd = (const float*)d_in[4];
  const float* W  = (const float*)d_in[5];
  const float* bias = (const float*)d_in[6];
  float* out = (float*)d_out;

  char* ws = (char*)d_ws;
  bf16* xbf = (bf16*)(ws);                           // 16 MB  [4096,2048]
  bf16* xT  = (bf16*)(ws + (16u << 20));             // 16 MB  [2048,4096]
  bf16* wbf = (bf16*)(ws + (32u << 20));             //  8 MB  [2048,2048]
  bf16* yT  = (bf16*)(ws + (40u << 20));             // 16 MB  [2048,4096]
  bf16* nwb = (bf16*)(ws + (56u << 20));             //  8 MB  [2048,2048]
  float* xm = (float*)(ws + (64u << 20));            //  8 KB
  float* ym = (float*)(ws + (64u << 20) + 8192u);    //  8 KB

  k_zero<<<8, 256, 0, stream>>>(xm, 2048);
  k_xprep<<<dim3(64, 128), dim3(32, 8), 0, stream>>>(x, xbf, xT);
  k_colsum<<<dim3(8, 32), 256, 0, stream>>>(x, xm);
  k_f2bf<<<2048, 256, 0, stream>>>(W, wbf);
  k_ym<<<2048, 256, 0, stream>>>(W, xm, ym);

  // GEMM1: y_^T (bf16): M=4096(B), N=2048(OUT), K=2048(IN)
  k_gemm<0><<<dim3(16, 16), 256, 0, stream>>>(
      xbf, wbf, 4096, 2048, 2048, yT, nullptr, nullptr, nullptr, nullptr,
      nullptr, nullptr, nullptr, nullptr, nullptr, nullptr);

  // GEMM2 + fused update: C = xT @ y_ (== G^T): M=2048(IN), N=2048(OUT), K=4096(B)
  k_gemm<1><<<dim3(16, 8), 256, 0, stream>>>(
      xT, yT, 2048, 2048, 4096, nullptr, W, wa, wb, wc, wd, xm, ym, nwb,
      nullptr, nullptr);

  // GEMM3: out = x @ new_W^T + bias: M=4096, N=2048, K=2048
  k_gemm<2><<<dim3(16, 16), 256, 0, stream>>>(
      xbf, nwb, 4096, 2048, 2048, nullptr, nullptr, nullptr, nullptr, nullptr,
      nullptr, nullptr, nullptr, nullptr, bias, out);
}
